// SelfAttentionVarSizedElementReduce_50302656971015
// MI455X (gfx1250) — matrix operand order repaired and checked by reading, not compiled, not run on hardware
//
#include <hip/hip_runtime.h>

typedef _Float16 v16h  __attribute__((ext_vector_type(16)));
typedef _Float16 half8 __attribute__((ext_vector_type(8)));
typedef _Float16 half4 __attribute__((ext_vector_type(4)));
typedef float    v8f   __attribute__((ext_vector_type(8)));
typedef float    f4v   __attribute__((ext_vector_type(4)));

#define D 128

// ---------------------------------------------------------------------------
// K0: fold Wq,Wk into Ct[d][e] = sum_h Wk[h,d]*Wq[h,e]  (so qW = mean @ C,
// with the B operand needing rows indexed by output column). Also convert Wo
// to f16 (Wo[n][k] is already in B-operand row layout for out = y @ Wo^T).
// ---------------------------------------------------------------------------
__global__ void prep_kernel(const float* __restrict__ Wq, const float* __restrict__ Wk,
                            const float* __restrict__ Wo,
                            _Float16* __restrict__ Ct, _Float16* __restrict__ Wo16) {
    int t = blockIdx.x * blockDim.x + threadIdx.x;
    if (t >= D * D) return;
    int d = t >> 7, e = t & 127;
    float s = 0.f;
    #pragma unroll 4
    for (int h = 0; h < D; ++h)
        s = fmaf(Wk[h * D + d], Wq[h * D + e], s);
    Ct[t]   = (_Float16)s;
    Wo16[t] = (_Float16)Wo[t];
}

__device__ __forceinline__ int lower_bound(const int* __restrict__ idx, int V, int key) {
    int lo = 0, hi = V;
    while (lo < hi) {
        int mid = (lo + hi) >> 1;
        if (idx[mid] < key) lo = mid + 1; else hi = mid;
    }
    return lo;
}

// ---------------------------------------------------------------------------
// K1: per-segment mean of x (one wave32 per segment; lane owns 4 columns).
// Also records segment [start,count) for the attention kernel.
// ---------------------------------------------------------------------------
__global__ void seg_mean_kernel(const float* __restrict__ x, const int* __restrict__ idx,
                                int V, int S, _Float16* __restrict__ m16,
                                int* __restrict__ segStart, int* __restrict__ segCnt) {
    int wave = (blockIdx.x * blockDim.x + threadIdx.x) >> 5;
    int lane = threadIdx.x & 31;
    if (wave >= S) return;
    int start = lower_bound(idx, V, wave);
    int end   = lower_bound(idx, V, wave + 1);
    int cnt   = end - start;
    if (lane == 0) { segStart[wave] = start; segCnt[wave] = cnt; }
    f4v acc = {0.f, 0.f, 0.f, 0.f};
    for (int v = start; v < end; ++v)
        acc += *(const f4v*)(x + (size_t)v * D + lane * 4);
    float inv = cnt > 0 ? 1.0f / (float)cnt : 0.0f;
    half4 o;
    o[0] = (_Float16)(acc[0] * inv); o[1] = (_Float16)(acc[1] * inv);
    o[2] = (_Float16)(acc[2] * inv); o[3] = (_Float16)(acc[3] * inv);
    *(half4*)(m16 + (size_t)wave * D + lane * 4) = o;
}

// ---------------------------------------------------------------------------
// WMMA GEMM: Out[M,128] = A[M,128](f16) x B^T, where B is stored so that row n
// holds the K-strip for output column n (B-operand layout). One wave per
// 16x128 output strip; A regs loaded once and reused across 8 N-tiles;
// K=128 -> 4 chained v_wmma_f32_16x16x32_f16 per tile.
// Lane layouts used here (D layout as in CDNA5 ISA):
//   A lane(h=L/16,r=L%16): 16 contiguous K at kc+16h of row m=r
//   B lane: 16 contiguous K at kc+16h of row n=r (same order as A; not the ISA table order)
//   D VGPR rr: row 8h+rr, col r
// ---------------------------------------------------------------------------
#define GEMM_WAVES 8
__global__ void wmma_gemm_kernel(const _Float16* __restrict__ A,
                                 const _Float16* __restrict__ B,
                                 float* __restrict__ Out, int Mtiles) {
    int mtile = blockIdx.x * GEMM_WAVES + (threadIdx.x >> 5);
    if (mtile >= Mtiles) return;                 // wave-uniform: EXEC stays all-1s
    int lane = threadIdx.x & 31;
    int h = lane >> 4;                            // 0 | 1
    int r = lane & 15;
    const _Float16* Arow = A + (size_t)(mtile * 16 + r) * D;

    v16h a[4];
    #pragma unroll
    for (int kc = 0; kc < 4; ++kc) {
        half8 lo = *(const half8*)(Arow + kc * 32 + h * 16);
        half8 hi = *(const half8*)(Arow + kc * 32 + h * 16 + 8);
        #pragma unroll
        for (int i = 0; i < 8; ++i) { a[kc][i] = lo[i]; a[kc][8 + i] = hi[i]; }
    }

    #pragma unroll
    for (int nt = 0; nt < 8; ++nt) {
        const _Float16* Brow = B + (size_t)(nt * 16 + r) * D + h * 16;
        v8f c = {0.f, 0.f, 0.f, 0.f, 0.f, 0.f, 0.f, 0.f};
        #pragma unroll
        for (int kc = 0; kc < 4; ++kc) {
            v16h b = *(const v16h*)(Brow + kc * 32);
            c = __builtin_amdgcn_wmma_f32_16x16x32_f16(false, a[kc], false, b,
                                                       (short)0, c, false, false);
        }
        #pragma unroll
        for (int rr = 0; rr < 8; ++rr)
            Out[(size_t)(mtile * 16 + h * 8 + rr) * D + nt * 16 + r] = c[rr];
    }
}

// ---------------------------------------------------------------------------
// K3: per-segment online softmax + weighted sum (flash-style, single pass
// over x). One wave32 per segment. scores_v = x_v . qW[seg]; running
// (max, denom, acc) rescaled as the max grows. y = acc / denom (f16).
// ---------------------------------------------------------------------------
__global__ void attn_kernel(const float* __restrict__ x, const float* __restrict__ qW,
                            const int* __restrict__ segStart, const int* __restrict__ segCnt,
                            int S, _Float16* __restrict__ y16) {
    int wave = (blockIdx.x * blockDim.x + threadIdx.x) >> 5;
    int lane = threadIdx.x & 31;
    if (wave >= S) return;
    int start = segStart[wave];
    int cnt   = segCnt[wave];
    f4v qw = *(const f4v*)(qW + (size_t)wave * D + lane * 4);

    float M = -3.402823466e38f;
    float denom = 0.f;
    f4v acc = {0.f, 0.f, 0.f, 0.f};
    for (int j = 0; j < cnt; ++j) {
        f4v xv = *(const f4v*)(x + (size_t)(start + j) * D + lane * 4);
        float d = fmaf(xv[0], qw[0], fmaf(xv[1], qw[1], fmaf(xv[2], qw[2], xv[3] * qw[3])));
        #pragma unroll
        for (int off = 16; off > 0; off >>= 1) d += __shfl_xor(d, off, 32);
        float nM = fmaxf(M, d);
        float sc = __expf(M - nM);    // exp(-huge) -> 0 on first iteration
        float p  = __expf(d - nM);
        denom = denom * sc + p;
        acc   = acc * sc + xv * p;
        M = nM;
    }
    float w = cnt > 0 ? 1.0f / denom : 0.f;
    half4 o;
    o[0] = (_Float16)(acc[0] * w); o[1] = (_Float16)(acc[1] * w);
    o[2] = (_Float16)(acc[2] * w); o[3] = (_Float16)(acc[3] * w);
    *(half4*)(y16 + (size_t)wave * D + lane * 4) = o;
}

// ---------------------------------------------------------------------------
// Launch: prep -> seg_mean -> GEMM(qW = m @ C) -> attn -> GEMM(out = y @ Wo^T)
// ---------------------------------------------------------------------------
extern "C" void kernel_launch(void* const* d_in, const int* in_sizes, int n_in,
                              void* d_out, int out_size, void* d_ws, size_t ws_size,
                              hipStream_t stream) {
    const float* x  = (const float*)d_in[0];
    const int*   idx = (const int*)d_in[1];
    const float* Wq = (const float*)d_in[3];
    const float* Wk = (const float*)d_in[4];
    const float* Wo = (const float*)d_in[5];
    const int V = in_sizes[1];
    const int S = out_size / D;

    char* ws = (char*)d_ws;
    size_t off = 0;
    auto alloc = [&](size_t bytes) -> void* {
        void* p = ws + off;
        off += (bytes + 255) & ~(size_t)255;
        return p;
    };
    _Float16* m16  = (_Float16*)alloc((size_t)S * D * sizeof(_Float16));
    _Float16* y16  = (_Float16*)alloc((size_t)S * D * sizeof(_Float16));
    float*    qW   = (float*)   alloc((size_t)S * D * sizeof(float));
    _Float16* Ct   = (_Float16*)alloc((size_t)D * D * sizeof(_Float16));
    _Float16* Wo16 = (_Float16*)alloc((size_t)D * D * sizeof(_Float16));
    int*      segStart = (int*) alloc((size_t)S * sizeof(int));
    int*      segCnt   = (int*) alloc((size_t)S * sizeof(int));

    prep_kernel<<<(D * D + 255) / 256, 256, 0, stream>>>(Wq, Wk, Wo, Ct, Wo16);

    int segBlocks = (S + 7) / 8;                 // 8 waves (segments) per block
    seg_mean_kernel<<<segBlocks, 256, 0, stream>>>(x, idx, V, S, m16, segStart, segCnt);

    int Mtiles = (S + 15) / 16;                  // S=50000 -> 3125 exact
    int gemmBlocks = (Mtiles + GEMM_WAVES - 1) / GEMM_WAVES;
    wmma_gemm_kernel<<<gemmBlocks, 32 * GEMM_WAVES, 0, stream>>>(m16, Ct, qW, Mtiles);

    attn_kernel<<<segBlocks, 256, 0, stream>>>(x, qW, segStart, segCnt, S, y16);

    wmma_gemm_kernel<<<gemmBlocks, 32 * GEMM_WAVES, 0, stream>>>(y16, Wo16, (float*)d_out, Mtiles);
}
